// RandLANetModel_68745246540413
// MI455X (gfx1250) — matrix operand order repaired and checked by reading, not compiled, not run on hardware
//
#include <hip/hip_runtime.h>
#include <hip/hip_bf16.h>
#include <math.h>

typedef __attribute__((ext_vector_type(16))) _Float16 v16h;
typedef __attribute__((ext_vector_type(8)))  float    v8f;

#define BN_INV 0.9999995000003749f  /* 1/sqrt(1+1e-6) */

// ---------------------------------------------------------------------------
// Weight packing: W (O x C, f32 row-major) -> Wp f16 in exact A-fragment order,
// zero-padded to Opad x (ktiles*32). Element for (ktile, M, half, e) holds
// W[M][ktile*32 + (e&7) + 16*(e>>3) + 8*half].
// ---------------------------------------------------------------------------
__global__ void pack_w_kernel(const float* __restrict__ W, _Float16* __restrict__ Wp,
                              int O, int C, int Opad, int ktiles)
{
    long t = (long)blockIdx.x * blockDim.x + threadIdx.x;
    long total = (long)Opad * ktiles * 32;
    if (t >= total) return;
    int e  = (int)(t & 15);
    int hf = (int)((t >> 4) & 1);
    int M  = (int)((t >> 5) % Opad);
    int kt = (int)(t / (32L * Opad));
    int K  = kt * 32 + (e & 7) + ((e >> 3) << 4) + (hf << 3);
    float v = (M < O && K < C) ? W[(long)M * C + K] : 0.0f;
    Wp[((long)kt * Opad + M) * 32 + hf * 16 + e] = (_Float16)v;
}

// ---------------------------------------------------------------------------
// Fused GEMM: Y[b,o,p] = act( BN( sum_c W[o,c]*X[b,c,p] + bias[o] ) + R[b,o,p] )
// One wave computes a 16(O) x 32(P) tile: one packed-f16 A fragment feeds two
// v_wmma_f32_16x16x32_f16. P % 32 == 0 by construction; K padded with zero
// weights so X is read unconditionally over Cpad rows.
// ---------------------------------------------------------------------------
__global__ __launch_bounds__(32)
void wmma_conv_kernel(const float* __restrict__ X, long xbs, int ldx,
                      const _Float16* __restrict__ Wp, int Opad,
                      const float* __restrict__ bias,
                      const float* __restrict__ g, const float* __restrict__ bb,
                      const float* __restrict__ R, long rbs, int ldr,
                      float* __restrict__ Y, long ybs, int ldy,
                      int Cpad, int O, int P, int act, float slope)
{
    const int tp   = blockIdx.x * 32;
    const int to   = blockIdx.y * 16;
    const int b    = blockIdx.z;
    const int lq   = threadIdx.x & 15;
    const int half = threadIdx.x >> 4;

    const float* Xb = X + (long)b * xbs;
    const int M  = to + lq;        // A-fragment row (output channel)
    const int N0 = tp + lq;        // B-fragment columns (positions)

    v8f acc0 = {}, acc1 = {};
    for (int k0 = 0; k0 < Cpad; k0 += 32) {
        v16h afrag = *(const v16h*)(Wp + (((long)(k0 >> 5) * Opad + M) << 5) + (half << 4));
        v16h b0, b1;
        const float* xp = Xb + (long)(k0 + (half << 3)) * ldx;
#pragma unroll
        for (int e = 0; e < 16; ++e) {
            const float* xr = xp + (long)((e & 7) + ((e >> 3) << 4)) * ldx;
            b0[e] = (_Float16)xr[N0];
            b1[e] = (_Float16)xr[N0 + 16];
        }
        acc0 = __builtin_amdgcn_wmma_f32_16x16x32_f16(
                   false, afrag, false, b0, (short)0, acc0, false, false);
        acc1 = __builtin_amdgcn_wmma_f32_16x16x32_f16(
                   false, afrag, false, b1, (short)0, acc1, false, false);
    }

#pragma unroll
    for (int r = 0; r < 8; ++r) {
        int o = to + r + (half << 3);   // C/D layout: M = r + 8*half, N = lane&15
        if (o < O) {
            float bi = bias ? bias[o] : 0.0f;
            float sc = 1.0f, sh = 0.0f;
            if (g) { sc = g[o] * BN_INV; sh = bb[o]; }
            float v0 = acc0[r] + bi, v1 = acc1[r] + bi;
            if (g) { v0 = v0 * sc + sh; v1 = v1 * sc + sh; }
            if (R) {
                const float* rr = R + (long)b * rbs + (long)o * ldr;
                v0 += rr[N0]; v1 += rr[N0 + 16];
            }
            if (act == 1) {
                v0 = v0 > 0.0f ? v0 : 0.0f;
                v1 = v1 > 0.0f ? v1 : 0.0f;
            } else if (act == 2) {
                v0 = v0 > 0.0f ? v0 : v0 * slope;
                v1 = v1 > 0.0f ? v1 : v1 * slope;
            }
            float* yr = Y + (long)b * ybs + (long)o * ldy;
            yr[N0]      = v0;
            yr[N0 + 16] = v1;
        }
    }
}

// ---------------------------------------------------------------------------
// Brute-force 16-NN with LDS tiling; register-resident insertion sort.
// coords: (B, nmax, 3). idx/dist: (B, m, 16).
// ---------------------------------------------------------------------------
__global__ void knn16_kernel(const float* __restrict__ coords, int m, int nmax,
                             int* __restrict__ idx, float* __restrict__ dist)
{
    const int q = blockIdx.x * blockDim.x + threadIdx.x;
    const int b = blockIdx.y;
    const float* cb = coords + (long)b * nmax * 3;
    __shared__ float sx[256], sy[256], sz[256];

    bool valid = q < m;
    float qx = 0, qy = 0, qz = 0;
    if (valid) { qx = cb[q*3]; qy = cb[q*3+1]; qz = cb[q*3+2]; }

    float dk[16]; int ik[16];
#pragma unroll
    for (int j = 0; j < 16; ++j) { dk[j] = 3.4e38f; ik[j] = 0; }

    for (int s0 = 0; s0 < m; s0 += 256) {
        int tile = (m - s0) < 256 ? (m - s0) : 256;
        for (int t = threadIdx.x; t < tile; t += blockDim.x) {
            sx[t] = cb[(s0+t)*3]; sy[t] = cb[(s0+t)*3+1]; sz[t] = cb[(s0+t)*3+2];
        }
        __syncthreads();
        if (valid) {
            for (int t = 0; t < tile; ++t) {
                float dx = qx - sx[t], dy = qy - sy[t], dz = qz - sz[t];
                float d2 = dx*dx + dy*dy + dz*dz;
                int   id = s0 + t;
                if (d2 < dk[15]) {
#pragma unroll
                    for (int j = 0; j < 16; ++j) {
                        bool sm = d2 < dk[j];
                        float td = dk[j]; int ti = ik[j];
                        dk[j] = sm ? d2 : dk[j]; ik[j] = sm ? id : ik[j];
                        d2    = sm ? td : d2;    id    = sm ? ti : id;
                    }
                }
            }
        }
        __syncthreads();
    }
    if (valid) {
#pragma unroll
        for (int j = 0; j < 16; ++j) {
            idx [((long)b*m + q)*16 + j] = ik[j];
            dist[((long)b*m + q)*16 + j] = sqrtf(fmaxf(dk[j], 1e-12f));
        }
    }
}

// 1-NN (decoder upsampling): nbr[b, q] = argmin over ms supports
__global__ void knn1_kernel(const float* __restrict__ coords, int ms, int mq,
                            int nmax, int* __restrict__ nbr)
{
    const int q = blockIdx.x * blockDim.x + threadIdx.x;
    const int b = blockIdx.y;
    const float* cb = coords + (long)b * nmax * 3;
    __shared__ float sx[256], sy[256], sz[256];
    bool valid = q < mq;
    float qx = 0, qy = 0, qz = 0;
    if (valid) { qx = cb[q*3]; qy = cb[q*3+1]; qz = cb[q*3+2]; }
    float best = 3.4e38f; int bi = 0;
    for (int s0 = 0; s0 < ms; s0 += 256) {
        int tile = (ms - s0) < 256 ? (ms - s0) : 256;
        for (int t = threadIdx.x; t < tile; t += blockDim.x) {
            sx[t] = cb[(s0+t)*3]; sy[t] = cb[(s0+t)*3+1]; sz[t] = cb[(s0+t)*3+2];
        }
        __syncthreads();
        if (valid) {
            for (int t = 0; t < tile; ++t) {
                float dx = qx - sx[t], dy = qy - sy[t], dz = qz - sz[t];
                float d2 = dx*dx + dy*dy + dz*dz;
                if (d2 < best) { best = d2; bi = s0 + t; }
            }
        }
        __syncthreads();
    }
    if (valid) nbr[(long)b*mq + q] = bi;
}

// Build LSE 10-channel geometry features: (B, 10, n, 16), p = q*16+j
__global__ void build_feat10_kernel(const float* __restrict__ coords, int nmax,
                                    const int* __restrict__ idx,
                                    const float* __restrict__ dist,
                                    float* __restrict__ out, int n)
{
    long t = (long)blockIdx.x * blockDim.x + threadIdx.x;
    int  b = blockIdx.y;
    long tot = (long)n * 16;
    if (t >= tot) return;
    int q = (int)(t >> 4), j = (int)(t & 15);
    const float* cb = coords + (long)b * nmax * 3;
    int   id = idx [((long)b*n + q)*16 + j];
    float d  = dist[((long)b*n + q)*16 + j];
    float cx = cb[q*3], cy = cb[q*3+1], cz = cb[q*3+2];
    float nx = cb[(long)id*3], ny = cb[(long)id*3+1], nz = cb[(long)id*3+2];
    float* ob = out + (long)b * 10 * tot;
    ob[0*tot + t] = cx;      ob[1*tot + t] = cy;      ob[2*tot + t] = cz;
    ob[3*tot + t] = nx;      ob[4*tot + t] = ny;      ob[5*tot + t] = nz;
    ob[6*tot + t] = cx - nx; ob[7*tot + t] = cy - ny; ob[8*tot + t] = cz - nz;
    ob[9*tot + t] = d;
}

// Broadcast (B,C,n) -> (B,Ctot,n,16) at channel offset co (replicate over k)
__global__ void bcast16_kernel(const float* __restrict__ src, long sbs,
                               float* __restrict__ dst, long dbs, int co,
                               int C, int n)
{
    long t = (long)blockIdx.x * blockDim.x + threadIdx.x;
    int  b = blockIdx.y;
    long tot = (long)C * n * 16;
    if (t >= tot) return;
    long row = (long)n * 16;
    int  c = (int)(t / row);
    long p = t % row;
    dst[(long)b*dbs + (long)(co + c)*row + p] =
        src[(long)b*sbs + (long)c*n + (p >> 4)];
}

// Attention pool: out[b,c,q] = sum_j softmax_j(S[b,c,q,:]) * X[b,c,q,j]
__global__ void softpool_kernel(const float* __restrict__ S,
                                const float* __restrict__ X, long bs,
                                float* __restrict__ out, long obs, int C, int n)
{
    long t = (long)blockIdx.x * blockDim.x + threadIdx.x;
    int  b = blockIdx.y;
    if (t >= (long)C * n) return;
    int c = (int)(t / n), q = (int)(t % n);
    long base = (long)b*bs + ((long)c*n + q) * 16;
    float sv[16]; float mx = -3.4e38f;
#pragma unroll
    for (int j = 0; j < 16; ++j) { sv[j] = S[base + j]; mx = sv[j] > mx ? sv[j] : mx; }
    float sum = 0.0f;
#pragma unroll
    for (int j = 0; j < 16; ++j) { sv[j] = __expf(sv[j] - mx); sum += sv[j]; }
    float inv = 1.0f / sum, acc = 0.0f;
#pragma unroll
    for (int j = 0; j < 16; ++j) acc += sv[j] * inv * X[base + j];
    out[(long)b*obs + t] = acc;
}

// Gather positions through nbr into concat buffer channels [co, co+C)
__global__ void gather_kernel(const float* __restrict__ src, long sbs, int lds,
                              const int* __restrict__ nbr, int nbs,
                              float* __restrict__ dst, long dbs, int ldd, int co,
                              int C, int P)
{
    long t = (long)blockIdx.x * blockDim.x + threadIdx.x;
    int  b = blockIdx.y;
    if (t >= (long)C * P) return;
    int c = (int)(t / P), q = (int)(t % P);
    dst[(long)b*dbs + (long)(co + c)*ldd + q] =
        src[(long)b*sbs + (long)c*lds + nbr[(long)b*nbs + q]];
}

// Strided channel copy into concat buffer
__global__ void copy_kernel(const float* __restrict__ src, long sbs, int lds,
                            float* __restrict__ dst, long dbs, int ldd, int co,
                            int C, int P)
{
    long t = (long)blockIdx.x * blockDim.x + threadIdx.x;
    int  b = blockIdx.y;
    if (t >= (long)C * P) return;
    int c = (int)(t / P), q = (int)(t % P);
    dst[(long)b*dbs + (long)(co + c)*ldd + q] = src[(long)b*sbs + (long)c*lds + q];
}

// fc_start (+bn_start, leaky 0.2) at permuted positions; also extracts coords
__global__ void fc_start_kernel(const float* __restrict__ inp,   // (B,N,6)
                                const int* __restrict__ perm,
                                const float* __restrict__ W,     // (8,6)
                                const float* __restrict__ bias,
                                const float* __restrict__ g,
                                const float* __restrict__ bb,
                                float* __restrict__ x,           // (B,8,N)
                                float* __restrict__ coords,      // (B,N,3)
                                int N)
{
    int j = blockIdx.x * blockDim.x + threadIdx.x;
    int b = blockIdx.y;
    if (j >= N) return;
    int src = perm[j];
    const float* ip = inp + ((long)b*N + src) * 6;
    float in[6];
#pragma unroll
    for (int d = 0; d < 6; ++d) in[d] = ip[d];
    float* cp = coords + ((long)b*N + j) * 3;
    cp[0] = in[0]; cp[1] = in[1]; cp[2] = in[2];
#pragma unroll
    for (int o = 0; o < 8; ++o) {
        float v = bias[o];
#pragma unroll
        for (int d = 0; d < 6; ++d) v += W[o*6 + d] * in[d];
        v = v * (g[o] * BN_INV) + bb[o];
        v = v > 0.0f ? v : 0.2f * v;
        x[(long)b*8*N + (long)o*N + j] = v;
    }
}

// Inverse-permutation scatter + (B,C,N)->(B,N,C) transpose, C=13
__global__ void final_scatter_kernel(const float* __restrict__ x, // (B,13,N)
                                     const int* __restrict__ perm,
                                     float* __restrict__ out, int N)
{
    int i = blockIdx.x * blockDim.x + threadIdx.x;
    int b = blockIdx.y;
    if (i >= N) return;
    int j = perm[i];
#pragma unroll
    for (int o = 0; o < 13; ++o)
        out[((long)b*N + j)*13 + o] = x[(long)b*13*N + (long)o*N + i];
}

// ---------------------------------------------------------------------------
// Host orchestration
// ---------------------------------------------------------------------------
extern "C" void kernel_launch(void* const* d_in, const int* in_sizes, int n_in,
                              void* d_out, int out_size, void* d_ws, size_t ws_size,
                              hipStream_t stream)
{
    (void)in_sizes; (void)n_in; (void)out_size; (void)ws_size;
    const int N = 8192, B = 2;

    // ---- parse inputs (setup_inputs order; params in JAX pytree sorted-key order)
    int t = 0;
    auto nf = [&](void)->const float* { return (const float*)d_in[t++]; };
    struct CP { const float *W, *b, *bb, *g; };
    auto ncp  = [&](void)->CP { CP p; p.W = nf(); p.b = nf(); p.bb = nf(); p.g = nf(); return p; };
    auto ncp2 = [&](void)->CP { CP p; p.W = nf(); p.b = nf(); p.bb = nullptr; p.g = nullptr; return p; };

    const float* input = nf();
    const float* bn_start_bb = nf();
    const float* bn_start_g  = nf();
    CP dec[4]; for (int i = 0; i < 4; ++i) dec[i] = ncp();
    struct LFA { CP lse1, lse2; const float *m1W,*m1b,*m2W,*m2b,*p1Ws; CP p1mlp;
                 const float *p2Ws; CP p2mlp; CP shrt; };
    LFA enc[4];
    for (int i = 0; i < 4; ++i) {
        enc[i].lse1 = ncp();
        enc[i].lse2 = ncp();
        enc[i].m1W = nf(); enc[i].m1b = nf();
        enc[i].m2W = nf(); enc[i].m2b = nf();
        enc[i].p1Ws = nf(); enc[i].p1mlp = ncp();
        enc[i].p2Ws = nf(); enc[i].p2mlp = ncp();
        enc[i].shrt = ncp();
    }
    CP fe0 = ncp(), fe1 = ncp(), fe2 = ncp2();
    const float* fsW = nf(); const float* fsb = nf();
    const float* mlpW = nf(); const float* mlpb = nf();
    t++;                                   // mask (all true; unused)
    const int* perm = (const int*)d_in[t++];

    // ---- workspace layout (floats)
    float* ws = (float*)d_ws;
    float* coords = ws;                 // 2*8192*3
    float* bufA   = coords + 49152;     // 4.4M
    float* bufB   = bufA + 4400000;     // 4.4M
    float* bufC   = bufB + 4400000;     // 2.2M
    float* bufD   = bufC + 2200000;     // 2.7M (feat10 / final logits)
    float* st0    = bufD + 2700000;     // (B,32,8192)
    float* st1    = st0 + 524288;       // (B,128,2048)
    float* st2    = st1 + 524288;       // (B,256,512)
    float* st3    = st2 + 262144;       // (B,512,128)
    float* shortb = st3 + 131072;       // residual branch
    float* feat0  = shortb + 524288;    // (B,8,8192)
    float* kdist  = feat0 + 131072;     // (B,8192,16)
    int*   kidx   = (int*)(kdist + 262144);
    int*   nbr    = kidx + 262144;      // (B,8192)
    _Float16* warena = (_Float16*)(nbr + 16384);  // packed weights (~1.3M halves)
    _Float16* wbump  = warena;

    // conv helper: packs weights (fragment order, zero-padded) then runs GEMM.
    auto conv = [&](const float* X, long xbs, int ldx,
                    const float* W, const float* bias,
                    const float* g, const float* bb,
                    const float* R, long rbs, int ldr,
                    float* Y, long ybs, int ldy,
                    int C, int O, int P, int act, float slope) {
        int ktiles = (C + 31) / 32;
        int Opad   = ((O + 15) / 16) * 16;
        _Float16* Wp = wbump;
        long nelem = (long)ktiles * Opad * 32;
        wbump += nelem;
        pack_w_kernel<<<dim3((unsigned)((nelem + 255) / 256)), 256, 0, stream>>>(
            W, Wp, O, C, Opad, ktiles);
        dim3 grid(P / 32, Opad / 16, B);
        wmma_conv_kernel<<<grid, 32, 0, stream>>>(X, xbs, ldx, Wp, Opad,
                                                  bias, g, bb, R, rbs, ldr,
                                                  Y, ybs, ldy,
                                                  ktiles * 32, O, P, act, slope);
    };
    auto flat = [&](long total)->dim3 { return dim3((unsigned)((total + 255) / 256), B); };

    // ---- fc_start + bn_start + leaky(0.2), permuted; coords extraction
    fc_start_kernel<<<flat(N), 256, 0, stream>>>(input, perm, fsW, fsb,
                                                 bn_start_g, bn_start_bb,
                                                 feat0, coords, N);

    // ---- encoder
    const int douts[4] = {16, 64, 128, 256};
    const int ns[4]    = {8192, 2048, 512, 128};
    float* stbuf[4] = {st0, st1, st2, st3};
    const float* xin = feat0; long xbs = (long)8 * N; int ldxin = N; int din = 8;
    for (int L = 0; L < 4; ++L) {
        const int n = ns[L], dO = douts[L], h = dO / 2;
        const LFA& E = enc[L];
        long row = (long)n * 16;
        // mlp1: din -> h, leaky 0.2
        conv(xin, xbs, ldxin, E.m1W, E.m1b, nullptr, nullptr, nullptr, 0, 0,
             bufC, (long)h * n, n, din, h, n, 2, 0.2f);
        // knn-16 on first n permuted points
        knn16_kernel<<<flat(n), 256, 0, stream>>>(coords, n, N, kidx, kdist);
        // geometry features (B,10,n,16)
        build_feat10_kernel<<<flat(row), 256, 0, stream>>>(coords, N, kidx, kdist, bufD, n);
        // lse1 conv (bn+relu) -> bufA channels [0,h)
        conv(bufD, 10 * row, (int)row, E.lse1.W, E.lse1.b, E.lse1.g, E.lse1.bb,
             nullptr, 0, 0, bufA, (long)dO * row, (int)row, 10, h, (int)row, 1, 0.0f);
        // broadcast mlp1 features -> bufA channels [h,2h)
        bcast16_kernel<<<flat((long)h * row), 256, 0, stream>>>(bufC, (long)h*n, bufA, (long)dO*row, h, h, n);
        // attention 1: scores = Ws * x
        conv(bufA, (long)dO * row, (int)row, E.p1Ws, nullptr, nullptr, nullptr,
             nullptr, 0, 0, bufB, (long)dO * row, (int)row, dO, dO, (int)row, 0, 0.0f);
        softpool_kernel<<<flat((long)dO * n), 256, 0, stream>>>(bufB, bufA, (long)dO*row, bufC, (long)dO*n, dO, n);
        // pool1 mlp: dO -> h, bn+relu
        conv(bufC, (long)dO * n, n, E.p1mlp.W, E.p1mlp.b, E.p1mlp.g, E.p1mlp.bb,
             nullptr, 0, 0, bufB, (long)h * n, n, dO, h, n, 1, 0.0f);
        // lse2 conv (bn+relu) -> bufA channels [0,h); broadcast pool1 out -> [h,2h)
        conv(bufD, 10 * row, (int)row, E.lse2.W, E.lse2.b, E.lse2.g, E.lse2.bb,
             nullptr, 0, 0, bufA, (long)dO * row, (int)row, 10, h, (int)row, 1, 0.0f);
        bcast16_kernel<<<flat((long)h * row), 256, 0, stream>>>(bufB, (long)h*n, bufA, (long)dO*row, h, h, n);
        // attention 2
        conv(bufA, (long)dO * row, (int)row, E.p2Ws, nullptr, nullptr, nullptr,
             nullptr, 0, 0, bufB, (long)dO * row, (int)row, dO, dO, (int)row, 0, 0.0f);
        softpool_kernel<<<flat((long)dO * n), 256, 0, stream>>>(bufB, bufA, (long)dO*row, bufC, (long)dO*n, dO, n);
        // pool2 mlp: dO -> dO, bn+relu
        conv(bufC, (long)dO * n, n, E.p2mlp.W, E.p2mlp.b, E.p2mlp.g, E.p2mlp.bb,
             nullptr, 0, 0, bufB, (long)dO * n, n, dO, dO, n, 1, 0.0f);
        // shortcut: din -> 2dO, bn, no act
        conv(xin, xbs, ldxin, E.shrt.W, E.shrt.b, E.shrt.g, E.shrt.bb,
             nullptr, 0, 0, shortb, (long)2*dO*n, n, din, 2*dO, n, 0, 0.0f);
        // mlp2 + residual, leaky 0.01 -> stack buffer
        conv(bufB, (long)dO * n, n, E.m2W, E.m2b, nullptr, nullptr,
             shortb, (long)2*dO*n, n, stbuf[L], (long)2*dO*n, n, dO, 2*dO, n, 2, 0.01f);
        xin = stbuf[L]; xbs = (long)2*dO*n; ldxin = n; din = 2*dO;
    }

    // ---- mid MLP 512->512 relu on first 32 positions (ld stays 128)
    conv(st3, (long)512*128, 128, mlpW, mlpb, nullptr, nullptr, nullptr, 0, 0,
         bufC, (long)512*32, 32, 512, 512, 32, 1, 0.0f);

    // ---- decoder
    const int msA[4] = {32, 128, 512, 2048};
    const int mqA[4] = {128, 512, 2048, 8192};
    const int CsA[4] = {512, 256, 128, 32};
    const int CoA[4] = {256, 128, 32, 8};
    float* skips[4] = {st3, st2, st1, st0};
    float* outs[4]  = {bufB, bufC, bufB, bufC};
    float* xsrc = bufC; int xl = 32, xc = 512;
    for (int L = 0; L < 4; ++L) {
        int ms = msA[L], mq = mqA[L], Cs = CsA[L], Co = CoA[L];
        int Cin = xc + Cs;
        knn1_kernel<<<flat(mq), 256, 0, stream>>>(coords, ms, mq, N, nbr);
        gather_kernel<<<flat((long)xc * mq), 256, 0, stream>>>(
            xsrc, (long)xc * xl, xl, nbr, mq, bufA, (long)Cin * mq, mq, 0, xc, mq);
        copy_kernel<<<flat((long)Cs * mq), 256, 0, stream>>>(
            skips[L], (long)Cs * mq, mq, bufA, (long)Cin * mq, mq, xc, Cs, mq);
        conv(bufA, (long)Cin * mq, mq, dec[L].W, dec[L].b, dec[L].g, dec[L].bb,
             nullptr, 0, 0, outs[L], (long)Co * mq, mq, Cin, Co, mq, 1, 0.0f);
        xsrc = outs[L]; xl = mq; xc = Co;
    }

    // ---- head: fc_end (8->64->32->13), then inverse-perm scatter + transpose
    conv(bufC, (long)8*N, N, fe0.W, fe0.b, fe0.g, fe0.bb, nullptr, 0, 0,
         bufA, (long)64*N, N, 8, 64, N, 1, 0.0f);
    conv(bufA, (long)64*N, N, fe1.W, fe1.b, fe1.g, fe1.bb, nullptr, 0, 0,
         bufB, (long)32*N, N, 64, 32, N, 1, 0.0f);
    conv(bufB, (long)32*N, N, fe2.W, fe2.b, nullptr, nullptr, nullptr, 0, 0,
         bufD, (long)13*N, N, 32, 13, N, 0, 0.0f);
    final_scatter_kernel<<<flat(N), 256, 0, stream>>>(bufD, perm, (float*)d_out, N);
}
